// TextEncoder_23545010716754
// MI455X (gfx1250) — matrix operand order repaired and checked by reading, not compiled, not run on hardware
//
#include <hip/hip_runtime.h>

typedef __bf16 bf16_t;
typedef __attribute__((ext_vector_type(16))) __bf16 v16bf;
typedef __attribute__((ext_vector_type(8)))  float  v8f;

#define NL   6
#define CCH  768
#define FCH  3072
#define NH   12
#define KCH  64
#define TT   512
#define BB   8
#define WREL 4

union V8 {
  uint4  v;
  bf16_t h[8];
};

// ---------------------------------------------------------------------------
// Elementwise helper kernels
// ---------------------------------------------------------------------------
__global__ void k_cast_bf16(const float* __restrict__ s, bf16_t* __restrict__ d, long n) {
  long i = (long)blockIdx.x * blockDim.x + threadIdx.x;
  long stride = (long)gridDim.x * blockDim.x;
  for (; i < n; i += stride) d[i] = (bf16_t)s[i];
}

// FFN conv weight recast + relayout: [NL][Mo][Ki][3] f32 -> [NL][3][Mo][Ki] bf16
__global__ void k_cast_w3(const float* __restrict__ s, bf16_t* __restrict__ d,
                          int Mo, int Ki) {
  long n = (long)NL * Mo * Ki * 3;
  long i = (long)blockIdx.x * blockDim.x + threadIdx.x;
  long stride = (long)gridDim.x * blockDim.x;
  for (; i < n; i += stride) {
    int k = (int)(i % Ki);
    long r = i / Ki;
    int m = (int)(r % Mo); r /= Mo;
    int tap = (int)(r % 3);
    int l = (int)(r / 3);
    d[i] = (bf16_t)s[(((long)l * Mo + m) * Ki + k) * 3 + tap];
  }
}

// emb_rel_{k,v}: [NL][1][9][KC] f32 -> padded [NL][16][KC] bf16 (rows 9..15 zero)
__global__ void k_embpad(const float* __restrict__ e, bf16_t* __restrict__ d) {
  int idx = blockIdx.x * blockDim.x + threadIdx.x;
  if (idx >= NL * 16 * KCH) return;
  int dd = idx & (KCH - 1);
  int j  = (idx >> 6) & 15;
  int l  = idx >> 10;
  float v = 0.f;
  if (j < 2 * WREL + 1) v = e[((long)l * (2 * WREL + 1) + j) * KCH + dd];
  d[idx] = (bf16_t)v;
}

__global__ void k_maskcopy(const float* __restrict__ x, const int* __restrict__ lens,
                           float* __restrict__ o, int ch) {
  long n = (long)BB * ch * TT;
  long i = (long)blockIdx.x * blockDim.x + threadIdx.x;
  long stride = (long)gridDim.x * blockDim.x;
  for (; i < n; i += stride) {
    int t = (int)(i % TT);
    int b = (int)(i / ((long)ch * TT));
    o[i] = (t < lens[b]) ? x[i] : 0.f;
  }
}

__global__ void k_maskcast(const float* __restrict__ x, const int* __restrict__ lens,
                           bf16_t* __restrict__ o, int ch, int relu) {
  long n = (long)BB * ch * TT;
  long i = (long)blockIdx.x * blockDim.x + threadIdx.x;
  long stride = (long)gridDim.x * blockDim.x;
  for (; i < n; i += stride) {
    int t = (int)(i % TT);
    int b = (int)(i / ((long)ch * TT));
    float v = x[i];
    if (relu) v = fmaxf(v, 0.f);
    o[i] = (t < lens[b]) ? (bf16_t)v : (bf16_t)0.f;
  }
}

// ---------------------------------------------------------------------------
// Generic strided batched bf16 WMMA GEMM:
//   D[bid] (+)= alpha * ( sum_taps A(MxK) * Bshift(KxN) + bias )
// A[m,k]  = A  [bid*a_bs + m*a_rs + k*a_cs + tap*a_ts]
// B[k,n]  = Bm [bid*b_bs + k*b_rs + (n+shift)*b_cs]   (0 if n+shift out of [0,N))
// D[m,n]  = Dp [bid*d_bs + m*d_rs + n*d_cs]           (f32 or bf16)
// Block tile 128x64, 8 waves (4x2), wave tile 32x32 = 2x2 WMMA 16x16x32 bf16.
// Staging is vectorized (b128) when the operand is contiguous along K or M/N.
// ---------------------------------------------------------------------------
__global__ __launch_bounds__(256)
void k_gemm_wmma(const bf16_t* __restrict__ A, long a_bs, long a_rs, long a_cs, long a_ts,
                 const bf16_t* __restrict__ Bm, long b_bs, long b_rs, long b_cs,
                 void* __restrict__ Dp, long d_bs, long d_rs, long d_cs,
                 const float* __restrict__ bias, float alpha,
                 int M, int N, int K, int taps, int shift0,
                 int acc_flag, int d_bf16) {
  __shared__ __align__(16) bf16_t As[128][32];  // A tile, K contiguous
  __shared__ __align__(16) bf16_t Bs[64][32];   // B tile transposed: [n][k]

  const int tid  = threadIdx.x;
  const int lane = tid & 31;
  const int wave = tid >> 5;
  const int wm   = wave >> 1;       // 0..3
  const int wn   = wave & 1;        // 0..1
  const int hh   = lane >> 4;       // 0/1
  const int lrow = lane & 15;
  const int m_blk = blockIdx.x * 128;
  const int n_blk = blockIdx.y * 64;
  const int bid   = blockIdx.z;

  const bf16_t* Ab = A  + (long)bid * a_bs;
  const bf16_t* Bb = Bm + (long)bid * b_bs;

  v8f c[2][2];
#pragma unroll
  for (int i = 0; i < 2; ++i)
#pragma unroll
    for (int j = 0; j < 2; ++j)
#pragma unroll
      for (int r = 0; r < 8; ++r) c[i][j][r] = 0.f;

  if (acc_flag) {
#pragma unroll
    for (int sm = 0; sm < 2; ++sm)
#pragma unroll
      for (int sn = 0; sn < 2; ++sn)
#pragma unroll
        for (int r = 0; r < 8; ++r) {
          int m = m_blk + wm * 32 + sm * 16 + hh * 8 + r;
          int n = n_blk + wn * 32 + sn * 16 + lrow;
          if (m < M && n < N) {
            long off = (long)bid * d_bs + (long)m * d_rs + (long)n * d_cs;
            c[sm][sn][r] = d_bf16 ? (float)((const bf16_t*)Dp)[off]
                                  : ((const float*)Dp)[off];
          }
        }
  }

  for (int tap = 0; tap < taps; ++tap) {
    const int shift  = shift0 + tap;
    const long a_off = (long)tap * a_ts;
    for (int k0 = 0; k0 < K; k0 += 32) {
      __syncthreads();
      if (k0 + 32 < K) {  // prefetch next K tile into caches
        __builtin_prefetch((const void*)(Ab + (long)m_blk * a_rs +
                                         (long)(k0 + 32) * a_cs + a_off), 0, 3);
        __builtin_prefetch((const void*)(Bb + (long)(k0 + 32) * b_rs +
                                         (long)n_blk * b_cs), 0, 3);
      }

      // ---------------- Stage A tile (128 x 32) ----------------
      if (a_cs == 1) {
        // K-contiguous: b128 global load + b128 LDS store
#pragma unroll
        for (int i = tid; i < 128 * 4; i += 256) {
          int r = i >> 2, kv = (i & 3) << 3;
          int gm = m_blk + r, gk = k0 + kv;
          V8 t; t.v.x = 0u; t.v.y = 0u; t.v.z = 0u; t.v.w = 0u;
          const bf16_t* src = Ab + (long)gm * a_rs + gk + a_off;
          if (gm < M && gk + 8 <= K) {
            t.v = *(const uint4*)src;
          } else if (gm < M) {
#pragma unroll
            for (int e = 0; e < 8; ++e)
              if (gk + e < K) t.h[e] = src[e];
          }
          *(uint4*)&As[r][kv] = t.v;
        }
      } else if (a_rs == 1) {
        // M-contiguous: b128 global load along m + strided b16 LDS stores
#pragma unroll
        for (int i = tid; i < 32 * 16; i += 256) {
          int kk = i >> 4, mv = (i & 15) << 3;
          int gm = m_blk + mv, gk = k0 + kk;
          V8 t; t.v.x = 0u; t.v.y = 0u; t.v.z = 0u; t.v.w = 0u;
          const bf16_t* src = Ab + (long)gk * a_cs + gm + a_off;
          if (gk < K) {
            if (gm + 8 <= M) {
              t.v = *(const uint4*)src;
            } else {
#pragma unroll
              for (int e = 0; e < 8; ++e)
                if (gm + e < M) t.h[e] = src[e];
            }
          }
#pragma unroll
          for (int e = 0; e < 8; ++e) As[mv + e][kk] = t.h[e];
        }
      } else {
        // generic fallback (unused by current call sites)
        for (int i = tid; i < 128 * 32; i += 256) {
          int r = i >> 5, kk = i & 31;
          int gm = m_blk + r, gk = k0 + kk;
          bf16_t v = (bf16_t)0.f;
          if (gm < M && gk < K) v = Ab[(long)gm * a_rs + (long)gk * a_cs + a_off];
          As[r][kk] = v;
        }
      }

      // ---------------- Stage B tile (transposed, 64 x 32) ----------------
      if (b_rs == 1 && shift == 0) {
        // K-contiguous: b128 global load + b128 LDS store
#pragma unroll
        for (int i = tid; i < 64 * 4; i += 256) {
          int nn = i >> 2, kv = (i & 3) << 3;
          int gn = n_blk + nn, gk = k0 + kv;
          V8 t; t.v.x = 0u; t.v.y = 0u; t.v.z = 0u; t.v.w = 0u;
          const bf16_t* src = Bb + (long)gn * b_cs + gk;
          if (gn < N && gk + 8 <= K) {
            t.v = *(const uint4*)src;
          } else if (gn < N) {
#pragma unroll
            for (int e = 0; e < 8; ++e)
              if (gk + e < K) t.h[e] = src[e];
          }
          *(uint4*)&Bs[nn][kv] = t.v;
        }
      } else if (b_cs == 1) {
        // N-contiguous (possibly shifted conv tap)
#pragma unroll
        for (int i = tid; i < 32 * 8; i += 256) {
          int kk = i >> 3, nv = (i & 7) << 3;
          int gk = k0 + kk;
          int gn0 = n_blk + nv;
          int gs0 = gn0 + shift;
          V8 t; t.v.x = 0u; t.v.y = 0u; t.v.z = 0u; t.v.w = 0u;
          if (gk < K) {
            const bf16_t* src = Bb + (long)gk * b_rs + gs0;
            if (shift == 0 && gn0 + 8 <= N) {
              t.v = *(const uint4*)src;
            } else {
#pragma unroll
              for (int e = 0; e < 8; ++e) {
                int gn = gn0 + e, gs = gs0 + e;
                if (gn < N && gs >= 0 && gs < N) t.h[e] = src[e];
              }
            }
          }
#pragma unroll
          for (int e = 0; e < 8; ++e) Bs[nv + e][kk] = t.h[e];
        }
      } else {
        // generic fallback (unused by current call sites)
        for (int i = tid; i < 64 * 32; i += 256) {
          int nn = i >> 5, kk = i & 31;
          int gn = n_blk + nn;
          int gs = gn + shift;
          int gk = k0 + kk;
          bf16_t v = (bf16_t)0.f;
          if (gn < N && gs >= 0 && gs < N && gk < K)
            v = Bb[(long)gk * b_rs + (long)gs * b_cs];
          Bs[nn][kk] = v;
        }
      }
      __syncthreads();

      // Build fragments per ISA VGPR layouts (wave32, 16x16x32 bf16)
      v16bf af[2], bfr[2];
#pragma unroll
      for (int sm = 0; sm < 2; ++sm) {
        const bf16_t* ap = &As[wm * 32 + sm * 16 + lrow][0];
#pragma unroll
        for (int vv = 0; vv < 8; ++vv) {
          int kk = 2 * vv + ((vv & 4) ? 8 : 0) + hh * 8;
          af[sm][2 * vv]     = ap[kk];
          af[sm][2 * vv + 1] = ap[kk + 1];
        }
      }
#pragma unroll
      for (int sn = 0; sn < 2; ++sn) {
        const bf16_t* bp = &Bs[wn * 32 + sn * 16 + lrow][0];
#pragma unroll
        for (int vv = 0; vv < 8; ++vv) {
          int kk = 2 * vv + ((vv & 4) ? 8 : 0) + hh * 8;
          bfr[sn][2 * vv]     = bp[kk];
          bfr[sn][2 * vv + 1] = bp[kk + 1];
        }
      }
#pragma unroll
      for (int sm = 0; sm < 2; ++sm)
#pragma unroll
        for (int sn = 0; sn < 2; ++sn)
          c[sm][sn] = __builtin_amdgcn_wmma_f32_16x16x32_bf16(
              false, af[sm], false, bfr[sn], (short)0, c[sm][sn], false, false);
    }
  }

  // Store D tile
#pragma unroll
  for (int sm = 0; sm < 2; ++sm)
#pragma unroll
    for (int sn = 0; sn < 2; ++sn)
#pragma unroll
      for (int r = 0; r < 8; ++r) {
        int m = m_blk + wm * 32 + sm * 16 + hh * 8 + r;
        int n = n_blk + wn * 32 + sn * 16 + lrow;
        if (m < M && n < N) {
          float val = c[sm][sn][r];
          if (!acc_flag) {
            if (bias) val += bias[m];
            val *= alpha;
          }
          long off = (long)bid * d_bs + (long)m * d_rs + (long)n * d_cs;
          if (d_bf16) ((bf16_t*)Dp)[off] = (bf16_t)val;
          else        ((float*)Dp)[off]  = val;
        }
      }
}

// ---------------------------------------------------------------------------
// Fused: scores + banded rel logits + mask + softmax -> P (bf16)
// and band extraction BW[t][j] = P[t][t+j-4] (abs_to_rel restricted to the
// 9 nonzero rel_v rows).  One block per (b,h,t) row of length T.
// ---------------------------------------------------------------------------
__global__ __launch_bounds__(256)
void k_softmax(const float* __restrict__ SC, const float* __restrict__ BL,
               const int* __restrict__ lens,
               bf16_t* __restrict__ P, bf16_t* __restrict__ BW) {
  const int t = blockIdx.x;
  const int h = blockIdx.y;
  const int b = blockIdx.z;
  const int tid = threadIdx.x;
  const long bh = (long)b * NH + h;
  const long base = (bh * TT + t) * TT;
  const long blbase = (bh * TT + t) * 16;
  const int len = lens[b];

  __shared__ float row[TT];
  __shared__ float red[256];

#pragma unroll
  for (int it = 0; it < 2; ++it) {
    int s = tid + it * 256;
    float v = SC[base + s];
    int j = s - t + WREL;
    if (j >= 0 && j < 2 * WREL + 1) v += BL[blbase + j];
    bool valid = (s < len) && (t < len);
    row[s] = valid ? v : -1e4f;
  }
  __syncthreads();
  red[tid] = fmaxf(row[tid], row[tid + 256]);
  __syncthreads();
  for (int off = 128; off > 0; off >>= 1) {
    if (tid < off) red[tid] = fmaxf(red[tid], red[tid + off]);
    __syncthreads();
  }
  float rmax = red[0];
  __syncthreads();
  float e0 = __expf(row[tid] - rmax);
  float e1 = __expf(row[tid + 256] - rmax);
  row[tid] = e0; row[tid + 256] = e1;
  red[tid] = e0 + e1;
  __syncthreads();
  for (int off = 128; off > 0; off >>= 1) {
    if (tid < off) red[tid] += red[tid + off];
    __syncthreads();
  }
  float inv = 1.f / red[0];
  __syncthreads();
  float p0 = row[tid] * inv, p1 = row[tid + 256] * inv;
  row[tid] = p0; row[tid + 256] = p1;
  P[base + tid]       = (bf16_t)p0;
  P[base + tid + 256] = (bf16_t)p1;
  __syncthreads();
  if (tid < 16) {
    float v = 0.f;
    int s = t + tid - WREL;
    if (tid < 2 * WREL + 1 && s >= 0 && s < TT) v = row[s];
    BW[blbase + tid] = (bf16_t)v;
  }
}

// ---------------------------------------------------------------------------
// x = LayerNorm_channel(x + (mask_y ? y*mask : y)) * g + b;   one block per (t,b)
// ---------------------------------------------------------------------------
__global__ __launch_bounds__(256)
void k_layernorm(float* __restrict__ X, const float* __restrict__ Y,
                 const float* __restrict__ g, const float* __restrict__ be,
                 const int* __restrict__ lens, int mask_y) {
  const int t = blockIdx.x;
  const int b = blockIdx.y;
  const int tid = threadIdx.x;
  const long base = (long)b * CCH * TT + t;
  const int len = lens[b];
  __shared__ float red[256];
  float vals[3];
  float s = 0.f;
#pragma unroll
  for (int i = 0; i < 3; ++i) {
    int ch = tid * 3 + i;
    float yv = Y[base + (long)ch * TT];
    if (mask_y && t >= len) yv = 0.f;
    float v = X[base + (long)ch * TT] + yv;
    vals[i] = v; s += v;
  }
  red[tid] = s; __syncthreads();
  for (int off = 128; off > 0; off >>= 1) {
    if (tid < off) red[tid] += red[tid + off];
    __syncthreads();
  }
  float mean = red[0] * (1.f / CCH);
  __syncthreads();
  float vs = 0.f;
#pragma unroll
  for (int i = 0; i < 3; ++i) { float d = vals[i] - mean; vs += d * d; }
  red[tid] = vs; __syncthreads();
  for (int off = 128; off > 0; off >>= 1) {
    if (tid < off) red[tid] += red[tid + off];
    __syncthreads();
  }
  float rstd = rsqrtf(red[0] * (1.f / CCH) + 1e-5f);
  __syncthreads();
#pragma unroll
  for (int i = 0; i < 3; ++i) {
    int ch = tid * 3 + i;
    X[base + (long)ch * TT] = (vals[i] - mean) * rstd * g[ch] + be[ch];
  }
}

// ---------------------------------------------------------------------------
// Host side
// ---------------------------------------------------------------------------
static inline void launch_gemm(hipStream_t stream,
                               const bf16_t* A, long a_bs, long a_rs, long a_cs, long a_ts,
                               const bf16_t* B, long b_bs, long b_rs, long b_cs,
                               void* D, long d_bs, long d_rs, long d_cs,
                               const float* bias, float alpha,
                               int M, int N, int K, int taps, int shift0, int batch,
                               int acc_flag, int d_bf16) {
  dim3 grid((M + 127) / 128, (N + 63) / 64, batch);
  k_gemm_wmma<<<grid, 256, 0, stream>>>(A, a_bs, a_rs, a_cs, a_ts,
                                        B, b_bs, b_rs, b_cs,
                                        D, d_bs, d_rs, d_cs,
                                        bias, alpha, M, N, K, taps, shift0,
                                        acc_flag, d_bf16);
}

static inline int nblk(long n) {
  long b = (n + 255) / 256;
  if (b > 16384) b = 16384;
  return (int)b;
}

extern "C" void kernel_launch(void* const* d_in, const int* in_sizes, int n_in,
                              void* d_out, int out_size, void* d_ws, size_t ws_size,
                              hipStream_t stream) {
  (void)in_sizes; (void)n_in; (void)out_size; (void)ws_size;
  const float* x_in = (const float*)d_in[0];
  const int*   lens = (const int*)d_in[1];
  const float* cqw = (const float*)d_in[2];
  const float* cqb = (const float*)d_in[3];
  const float* ckw = (const float*)d_in[4];
  const float* ckb = (const float*)d_in[5];
  const float* cvw = (const float*)d_in[6];
  const float* cvb = (const float*)d_in[7];
  const float* cow = (const float*)d_in[8];
  const float* cob = (const float*)d_in[9];
  const float* erk = (const float*)d_in[10];
  const float* erv = (const float*)d_in[11];
  const float* n1g = (const float*)d_in[12];
  const float* n1b = (const float*)d_in[13];
  const float* fw1 = (const float*)d_in[14];
  const float* fb1 = (const float*)d_in[15];
  const float* fw2 = (const float*)d_in[16];
  const float* fb2 = (const float*)d_in[17];
  const float* n2g = (const float*)d_in[18];
  const float* n2b = (const float*)d_in[19];

  char* wsp = (char*)d_ws;
  size_t cur = 0;
  auto alloc = [&](size_t bytes) -> void* {
    void* p = (void*)(wsp + cur);
    cur += (bytes + 255) & ~(size_t)255;
    return p;
  };

  const long BCT  = (long)BB * CCH * TT;
  const long BHTT = (long)BB * NH * TT * TT;
  const long BFT  = (long)BB * FCH * TT;
  const long CT   = (long)CCH * TT;
  const long KT   = (long)KCH * TT;

  float*  XW = (float*)alloc(BCT * 4);                 // running x (fp32)
  bf16_t* XM = (bf16_t*)alloc(BCT * 2);                // masked x bf16
  bf16_t* Qb = (bf16_t*)alloc(BCT * 2);
  bf16_t* Kb = (bf16_t*)alloc(BCT * 2);
  bf16_t* Vb = (bf16_t*)alloc(BCT * 2);
  float*  SC = (float*)alloc(BHTT * 4);                // attention scores
  float*  BL = (float*)alloc((long)BB * NH * TT * 16 * 4);  // band rel logits
  bf16_t* Pb = (bf16_t*)alloc(BHTT * 2);               // softmax probs bf16
  bf16_t* BW = (bf16_t*)alloc((long)BB * NH * TT * 16 * 2); // band weights
  bf16_t* AO = (bf16_t*)alloc(BCT * 2);                // attention output bf16
  float*  Yf = (float*)alloc(BCT * 4);                 // projection/ffn output
  float*  Hf = (float*)alloc(BFT * 4);                 // ffn hidden fp32
  bf16_t* Hb = (bf16_t*)alloc(BFT * 2);                // ffn hidden bf16
  bf16_t* Wq = (bf16_t*)alloc((long)NL * CCH * CCH * 2);
  bf16_t* Wk = (bf16_t*)alloc((long)NL * CCH * CCH * 2);
  bf16_t* Wv = (bf16_t*)alloc((long)NL * CCH * CCH * 2);
  bf16_t* Wo = (bf16_t*)alloc((long)NL * CCH * CCH * 2);
  bf16_t* W1 = (bf16_t*)alloc((long)NL * 3 * FCH * CCH * 2); // tap-major
  bf16_t* W2 = (bf16_t*)alloc((long)NL * 3 * CCH * FCH * 2); // tap-major
  bf16_t* EK = (bf16_t*)alloc((long)NL * 16 * KCH * 2);
  bf16_t* EV = (bf16_t*)alloc((long)NL * 16 * KCH * 2);

  // One-time weight casts (deterministic every call)
  auto cast = [&](const float* s, bf16_t* d, long n) {
    k_cast_bf16<<<nblk(n), 256, 0, stream>>>(s, d, n);
  };
  cast(cqw, Wq, (long)NL * CCH * CCH);
  cast(ckw, Wk, (long)NL * CCH * CCH);
  cast(cvw, Wv, (long)NL * CCH * CCH);
  cast(cow, Wo, (long)NL * CCH * CCH);
  k_cast_w3<<<nblk((long)NL * FCH * CCH * 3), 256, 0, stream>>>(fw1, W1, FCH, CCH);
  k_cast_w3<<<nblk((long)NL * CCH * FCH * 3), 256, 0, stream>>>(fw2, W2, CCH, FCH);
  k_embpad<<<(NL * 16 * KCH + 255) / 256, 256, 0, stream>>>(erk, EK);
  k_embpad<<<(NL * 16 * KCH + 255) / 256, 256, 0, stream>>>(erv, EV);

  // x = x * mask
  k_maskcopy<<<nblk(BCT), 256, 0, stream>>>(x_in, lens, XW, CCH);

  const float scale = 0.125f;  // KC^-0.5

  for (int l = 0; l < NL; ++l) {
    const bf16_t* wq = Wq + (long)l * CCH * CCH;
    const bf16_t* wk = Wk + (long)l * CCH * CCH;
    const bf16_t* wv = Wv + (long)l * CCH * CCH;
    const bf16_t* wo = Wo + (long)l * CCH * CCH;
    const bf16_t* w1 = W1 + (long)l * 3 * FCH * CCH;
    const bf16_t* w2 = W2 + (long)l * 3 * CCH * FCH;
    const bf16_t* ek = EK + (long)l * 16 * KCH;
    const bf16_t* ev = EV + (long)l * 16 * KCH;

    // xm = mask * x (bf16)
    k_maskcast<<<nblk(BCT), 256, 0, stream>>>(XW, lens, XM, CCH, 0);

    // q/k/v = (W @ xm + b) [* scale for q];  batch over B
    launch_gemm(stream, wq, 0, CCH, 1, 0, XM, CT, TT, 1, Qb, CT, TT, 1,
                cqb + (long)l * CCH, scale, CCH, TT, CCH, 1, 0, BB, 0, 1);
    launch_gemm(stream, wk, 0, CCH, 1, 0, XM, CT, TT, 1, Kb, CT, TT, 1,
                ckb + (long)l * CCH, 1.f, CCH, TT, CCH, 1, 0, BB, 0, 1);
    launch_gemm(stream, wv, 0, CCH, 1, 0, XM, CT, TT, 1, Vb, CT, TT, 1,
                cvb + (long)l * CCH, 1.f, CCH, TT, CCH, 1, 0, BB, 0, 1);

    // scores[bh] = Q^T K  (M=T, N=T, K=KC), batch over B*H
    launch_gemm(stream, Qb, KT, 1, TT, 0, Kb, KT, TT, 1,
                SC, (long)TT * TT, TT, 1, nullptr, 1.f,
                TT, TT, KCH, 1, 0, BB * NH, 0, 0);
    // band rel logits: Q @ emb_rel_k^T (only 9 nonzero rel rows, padded to 16)
    launch_gemm(stream, Qb, KT, 1, TT, 0, ek, 0, 1, KCH,
                BL, (long)TT * 16, 16, 1, nullptr, 1.f,
                TT, 16, KCH, 1, 0, BB * NH, 0, 0);

    // fused rel_to_abs + mask + softmax + abs_to_rel band extraction
    k_softmax<<<dim3(TT, NH, BB), 256, 0, stream>>>(SC, BL, lens, Pb, BW);

    // out = P @ V   (store into [B, C, T] layout as bf16)
    launch_gemm(stream, Pb, (long)TT * TT, TT, 1, 0, Vb, KT, 1, TT,
                AO, KT, 1, TT, nullptr, 1.f,
                TT, KCH, TT, 1, 0, BB * NH, 0, 1);
    // out += band_w @ emb_rel_v  (K=16, accumulate into AO)
    launch_gemm(stream, BW, (long)TT * 16, 16, 1, 0, ev, 0, KCH, 1,
                AO, KT, 1, TT, nullptr, 1.f,
                TT, KCH, 16, 1, 0, BB * NH, 1, 1);

    // y = Wo @ out + bo
    launch_gemm(stream, wo, 0, CCH, 1, 0, AO, CT, TT, 1, Yf, CT, TT, 1,
                cob + (long)l * CCH, 1.f, CCH, TT, CCH, 1, 0, BB, 0, 0);

    // x = LN(x + y)
    k_layernorm<<<dim3(TT, BB), 256, 0, stream>>>(
        XW, Yf, n1g + (long)l * CCH, n1b + (long)l * CCH, lens, 0);

    // FFN: h = relu(W1 *conv3* (x*mask) + b1)   (tap-major weights, K-contig A)
    k_maskcast<<<nblk(BCT), 256, 0, stream>>>(XW, lens, XM, CCH, 0);
    launch_gemm(stream, w1, 0, CCH, 1, (long)FCH * CCH, XM, CT, TT, 1,
                Hf, (long)FCH * TT, TT, 1, fb1 + (long)l * FCH, 1.f,
                FCH, TT, CCH, 3, -1, BB, 0, 0);
    k_maskcast<<<nblk(BFT), 256, 0, stream>>>(Hf, lens, Hb, FCH, 1);
    // y = W2 *conv3* (h*mask) + b2
    launch_gemm(stream, w2, 0, FCH, 1, (long)CCH * FCH, Hb, (long)FCH * TT, TT, 1,
                Yf, CT, TT, 1, fb2 + (long)l * CCH, 1.f,
                CCH, TT, FCH, 3, -1, BB, 0, 0);
    // x = LN(x + y*mask)
    k_layernorm<<<dim3(TT, BB), 256, 0, stream>>>(
        XW, Yf, n2g + (long)l * CCH, n2b + (long)l * CCH, lens, 1);
  }

  // out = x * mask (fp32)
  k_maskcopy<<<nblk(BCT), 256, 0, stream>>>(XW, lens, (float*)d_out, CCH);
}
